// EfficientCostVolumeManager_28432683500059
// MI455X (gfx1250) — matrix operand order repaired and checked by reading, not compiled, not run on hardware
//
#include <hip/hip_runtime.h>
#include <math.h>

typedef __attribute__((ext_vector_type(16))) _Float16 v16h;
typedef __attribute__((ext_vector_type(8)))  _Float16 v8h;
typedef __attribute__((ext_vector_type(8)))  float    v8f;

#define HH 96
#define WW 128
#define CC 16
#define KK 7
#define DD 64
#define DCHUNK 8
#define EPSF 1e-8f

// workspace layout (bytes)
#define WS_DEPTH_OFF 0                         // 64 floats
#define WS_MT_OFF    256                       // 7 * 12 floats (M3 row-major 9 + t 3)
#define WS_CUR_OFF   1024                      // H*W*16 halfs
#define WS_SRC_OFF   (1024 + HH*WW*CC*2)       // K*H*W*16 halfs

static __device__ __forceinline__ int iclampi(int v, int hi) {
    return v < 0 ? 0 : (v > hi ? hi : v);
}
static __device__ __forceinline__ v8h splat8(_Float16 x) {
    v8h r = {x, x, x, x, x, x, x, x};
    return r;
}

// ---------------------------------------------------------------------------
// Prep 1: per-view projection matrices M3 = (Ks*E)[:3,:3] * invK[:3,:3], t,
//         and log-spaced depth plane LUT.
// ---------------------------------------------------------------------------
__global__ void prep_params_kernel(const float* __restrict__ extr,
                                   const float* __restrict__ Ks,
                                   const float* __restrict__ invK,
                                   const float* __restrict__ mind,
                                   const float* __restrict__ maxd,
                                   float* __restrict__ ws_depth,
                                   float* __restrict__ ws_Mt) {
    int t = threadIdx.x;
    if (t < DD) {
        float r = (float)t / (float)(DD - 1);
        ws_depth[t] = __expf(__logf(mind[0]) + __logf(maxd[0] / mind[0]) * r);
    }
    if (t == 0) {
        for (int k = 0; k < KK; ++k) {
            const float* E  = extr + k * 16;
            const float* Km = Ks + k * 16;
            float P[3][4];
            for (int i = 0; i < 3; ++i)
                for (int j = 0; j < 4; ++j) {
                    float s = 0.f;
                    for (int l = 0; l < 4; ++l) s += Km[i * 4 + l] * E[l * 4 + j];
                    P[i][j] = s;
                }
            for (int i = 0; i < 3; ++i) {
                for (int j = 0; j < 3; ++j) {
                    float s = 0.f;
                    for (int l = 0; l < 3; ++l) s += P[i][l] * invK[l * 4 + j];
                    ws_Mt[k * 12 + i * 3 + j] = s;
                }
                ws_Mt[k * 12 + 9 + i] = P[i][3];
            }
        }
    }
}

// ---------------------------------------------------------------------------
// Prep 2: transpose + f16-convert features into channel-contiguous layout:
//   srcT[k][h][w][c], curT[h][w][c]  (32B per pixel, 16B-aligned halves)
// ---------------------------------------------------------------------------
__global__ void prep_feats_kernel(const float* __restrict__ cur,
                                  const float* __restrict__ src,
                                  _Float16* __restrict__ curT,
                                  _Float16* __restrict__ srcT) {
    int idx = blockIdx.x * blockDim.x + threadIdx.x;
    const int total = (KK + 1) * HH * WW * CC;
    if (idx >= total) return;
    int c = idx & (CC - 1);
    int p = idx >> 4;            // pixel index within slice
    int w = p % WW;
    int h = (p / WW) % HH;
    int s = idx / (CC * HH * WW);
    if (s < KK) {
        srcT[((s * HH + h) * WW + w) * CC + c] =
            (_Float16)src[((s * CC + c) * HH + h) * WW + w];
    } else {
        curT[(h * WW + w) * CC + c] = (_Float16)cur[(c * HH + h) * WW + w];
    }
}

// ---------------------------------------------------------------------------
// Main: one wave owns 16 pixels (a W-strip) for 8 depth planes.
//   A (16x32 f16): rows = pixels, K = 2 views x 16 channels (view pair)
//   B (32x16 f16): cur features, same 16-ch vector in both K halves
//   C chains over the 4 view pairs; diag(D) = cost[d, pixel].
// All view loops are fully unrolled so k is a compile-time constant
// (no dynamic register indexing / select chains).
// ---------------------------------------------------------------------------
__global__ __launch_bounds__(256) void cost_volume_wmma_kernel(
    const _Float16* __restrict__ srcT, const _Float16* __restrict__ curT,
    const float* __restrict__ Mt, const float* __restrict__ depths,
    float* __restrict__ out) {
    const int h      = blockIdx.x;          // 0..95
    const int dchunk = blockIdx.y;          // 0..7
    const int lane   = threadIdx.x & 31;
    const int wid    = threadIdx.x >> 5;    // wave id = W strip (0..7)
    const int pixn   = lane & 15;
    const int w      = wid * 16 + pixn;     // 0..127
    const int choff  = (lane < 16) ? 0 : 8; // which 8-channel half this lane gathers

    // B operand: this lane's 8-channel cur half (choff) of its pixel, repeated for both views.
    const _Float16* curp = curT + (h * WW + w) * CC;
    union { v16h v; v8h hv[2]; } B;
    B.hv[0] = *(const v8h*)(curp + choff);
    B.hv[1] = *(const v8h*)(curp + choff);

    // Hoist projection: a = M3 * (w+0.5, h+0.5, 1), per view.
    float ax[KK], ay[KK], az[KK], tx[KK], ty[KK], tz[KK];
    const float px = (float)w + 0.5f;
    const float py = (float)h + 0.5f;
#pragma unroll
    for (int k = 0; k < KK; ++k) {
        const float* M = Mt + k * 12;
        ax[k] = fmaf(M[0], px, fmaf(M[1], py, M[2]));
        ay[k] = fmaf(M[3], px, fmaf(M[4], py, M[5]));
        az[k] = fmaf(M[6], px, fmaf(M[7], py, M[8]));
        tx[k] = M[9]; ty[k] = M[10]; tz[k] = M[11];
    }

#pragma unroll 1
    for (int di = 0; di < DCHUNK; ++di) {
        const int d = dchunk * DCHUNK + di;
        const float dep = depths[d];
        v8f acc = {};

#pragma unroll
        for (int pr = 0; pr < 4; ++pr) {
            union { v16h v; v8h hv[2]; } A;
#pragma unroll
            for (int s = 0; s < 2; ++s) {
                const int k = pr * 2 + s;      // compile-time constant
                v8h wf = {};
                if (k < KK) {
                    float cz  = fmaf(dep, az[k], tz[k]);
                    float inv = __builtin_amdgcn_rcpf(cz + EPSF);
                    float sx  = fmaf(dep, ax[k], tx[k]) * inv - 0.5f;
                    float sy  = fmaf(dep, ay[k], ty[k]) * inv - 0.5f;
                    sx = fminf(fmaxf(sx, -8.0f), (float)(WW + 8));
                    sy = fminf(fmaxf(sy, -8.0f), (float)(HH + 8));
                    float fx0 = floorf(sx), fy0 = floorf(sy);
                    float wx1 = sx - fx0, wy1 = sy - fy0;
                    float wx0 = 1.0f - wx1, wy0 = 1.0f - wy1;
                    int x0 = (int)fx0, y0 = (int)fy0;
                    int x1 = x0 + 1, y1 = y0 + 1;
                    float mz = (cz > 0.0f) ? 1.0f : 0.0f;
                    bool vx0 = (x0 >= 0) & (x0 < WW);
                    bool vx1 = (x1 >= 0) & (x1 < WW);
                    bool vy0 = (y0 >= 0) & (y0 < HH);
                    bool vy1 = (y1 >= 0) & (y1 < HH);
                    int x0c = iclampi(x0, WW - 1), x1c = iclampi(x1, WW - 1);
                    int y0c = iclampi(y0, HH - 1), y1c = iclampi(y1, HH - 1);
                    const _Float16* kb = srcT + k * (HH * WW * CC) + choff;
                    v8h f00 = *(const v8h*)(kb + (y0c * WW + x0c) * CC);
                    v8h f01 = *(const v8h*)(kb + (y0c * WW + x1c) * CC);
                    v8h f10 = *(const v8h*)(kb + (y1c * WW + x0c) * CC);
                    v8h f11 = *(const v8h*)(kb + (y1c * WW + x1c) * CC);
                    float w00 = (vx0 & vy0) ? wx0 * wy0 * mz : 0.0f;
                    float w01 = (vx1 & vy0) ? wx1 * wy0 * mz : 0.0f;
                    float w10 = (vx0 & vy1) ? wx0 * wy1 * mz : 0.0f;
                    float w11 = (vx1 & vy1) ? wx1 * wy1 * mz : 0.0f;
                    wf  = f00 * splat8((_Float16)w00);
                    wf += f01 * splat8((_Float16)w01);
                    wf += f10 * splat8((_Float16)w10);
                    wf += f11 * splat8((_Float16)w11);
                }
                A.hv[s] = wf;
            }
            // D = A*B + C ; accumulate across view pairs in C.
            acc = __builtin_amdgcn_wmma_f32_16x16x32_f16(
                false, A.v, false, B.v, (short)0, acc, false, false);
        }

        // Diagonal extraction: diag element m lives at
        //   m<8  : vgpr m,   lane m
        //   m>=8 : vgpr m-8, lane m+16
        int sel = lane & 7;
        float v = acc[0];
        v = (sel == 1) ? acc[1] : v;
        v = (sel == 2) ? acc[2] : v;
        v = (sel == 3) ? acc[3] : v;
        v = (sel == 4) ? acc[4] : v;
        v = (sel == 5) ? acc[5] : v;
        v = (sel == 6) ? acc[6] : v;
        v = (sel == 7) ? acc[7] : v;
        bool havediag = (lane < 8) || (lane >= 24);
        int m = (lane < 8) ? lane : (lane - 16);
        if (havediag) {
            out[(d * HH + h) * WW + (wid * 16 + m)] = v;
        }
    }
}

// ---------------------------------------------------------------------------
extern "C" void kernel_launch(void* const* d_in, const int* in_sizes, int n_in,
                              void* d_out, int out_size, void* d_ws, size_t ws_size,
                              hipStream_t stream) {
    (void)in_sizes; (void)n_in; (void)out_size; (void)ws_size;
    const float* cur_feats = (const float*)d_in[0];  // [1,16,96,128]
    const float* src_feats = (const float*)d_in[1];  // [1,7,16,96,128]
    const float* extr      = (const float*)d_in[2];  // [1,7,4,4]
    // d_in[3] = src_poses (unused)
    const float* Ks        = (const float*)d_in[4];  // [1,7,4,4]
    const float* invK      = (const float*)d_in[5];  // [1,4,4]
    const float* mind      = (const float*)d_in[6];
    const float* maxd      = (const float*)d_in[7];
    float* out = (float*)d_out;

    char* ws = (char*)d_ws;
    float*    ws_depth = (float*)(ws + WS_DEPTH_OFF);
    float*    ws_Mt    = (float*)(ws + WS_MT_OFF);
    _Float16* curT     = (_Float16*)(ws + WS_CUR_OFF);
    _Float16* srcT     = (_Float16*)(ws + WS_SRC_OFF);

    prep_params_kernel<<<1, 64, 0, stream>>>(extr, Ks, invK, mind, maxd,
                                             ws_depth, ws_Mt);

    const int total = (KK + 1) * HH * WW * CC;
    prep_feats_kernel<<<(total + 255) / 256, 256, 0, stream>>>(cur_feats, src_feats,
                                                               curT, srcT);

    dim3 grid(HH, DD / DCHUNK, 1);   // 96 rows x 8 depth chunks
    cost_volume_wmma_kernel<<<grid, 256, 0, stream>>>(srcT, curT, ws_Mt, ws_depth, out);
}
